// NNSDecoder_83820581749529
// MI455X (gfx1250) — matrix operand order repaired and checked by reading, compiled, not run on hardware
//
#include <hip/hip_runtime.h>

#define Bsz 16
#define Ntot 501
#define Dd 128
#define Hh 4

typedef __attribute__((ext_vector_type(16))) _Float16 v16h;
typedef __attribute__((ext_vector_type(8))) float v8f;

// ---------------------------------------------------------------------------
// K1: fold compat projections.  For combo c in {pickup*set1, pickup*set2,
// delivery*set1, delivery*set2} and head h, compute
//   u[e]  = sum_d q[d] * Wq[h][d][e]
//   v[d'] = sum_e u[e] * Wk[h][d'][e]
// so that compat[b,n,h] = v . k[b,n].
// ---------------------------------------------------------------------------
__global__ __launch_bounds__(128) void k1_compat_vec(
    const float* __restrict__ h_hat, const int* __restrict__ pos_p,
    const int* __restrict__ pos_d,
    const float* __restrict__ Wq1, const float* __restrict__ Wk1,
    const float* __restrict__ Wq2, const float* __restrict__ Wk2,
    float* __restrict__ Vv) {
  int c = blockIdx.x >> 6;       // 4 combos, 64 (b,h) blocks each
  int rem = blockIdx.x & 63;
  int b = rem >> 2, h = rem & 3;
  int t = threadIdx.x;           // 0..127
  const float* Wq = (c & 1) ? Wq2 : Wq1;
  const float* Wk = (c & 1) ? Wk2 : Wk1;
  int pos = (c < 2) ? pos_p[b] : pos_d[b];

  __shared__ float sq[Dd];
  __shared__ float su[Dd];
  sq[t] = h_hat[((size_t)b * Ntot + pos) * Dd + t];
  __syncthreads();

  const float* wqh = Wq + h * Dd * Dd;
  float u = 0.f;
#pragma unroll 8
  for (int d = 0; d < Dd; ++d) u += sq[d] * wqh[d * Dd + t];
  su[t] = u;
  __syncthreads();

  const float* wkh = Wk + h * Dd * Dd + t * Dd;
  float v = 0.f;
#pragma unroll 8
  for (int e = 0; e < Dd; ++e) v += su[e] * wkh[e];
  Vv[((c * Bsz + b) * Hh + h) * Dd + t] = v;
}

// ---------------------------------------------------------------------------
// K2: one wave per (b,n).  16 dot-products of length 128 (vs h_hat row and
// gathered h_nb row), then fc1 split-projection, stored as f16:
//   Af[b,n,o] = sum_{k<8} cp[k]*fc1_w[k][o]
//   Bf[b,n,o] = sum_{k<8} cd[k]*fc1_w[8+k][o] + fc1_b[o]
// ---------------------------------------------------------------------------
__global__ __launch_bounds__(32) void k2_feat(
    const float* __restrict__ h_hat, const int* __restrict__ solution,
    const float* __restrict__ Vv, const float* __restrict__ fc1_w,
    const float* __restrict__ fc1_b,
    _Float16* __restrict__ Af, _Float16* __restrict__ Bf) {
  int bn = blockIdx.x;
  int b = bn / Ntot;
  int lane = threadIdx.x;
  int sol = solution[bn];

  float hh[4], hn[4];
  const float* hr = h_hat + (size_t)bn * Dd + lane;
  const float* nr = h_hat + ((size_t)b * Ntot + sol) * Dd + lane;
#pragma unroll
  for (int k = 0; k < 4; ++k) { hh[k] = hr[32 * k]; hn[k] = nr[32 * k]; }

  float cp[8], cd[8];
#pragma unroll
  for (int h = 0; h < 4; ++h) {
#pragma unroll
    for (int c = 0; c < 4; ++c) {
      const float* vv = Vv + (((c * Bsz) + b) * Hh + h) * Dd + lane;
      const float* x = (c & 1) ? hn : hh;
      float a = vv[0] * x[0] + vv[32] * x[1] + vv[64] * x[2] + vv[96] * x[3];
#pragma unroll
      for (int m = 1; m < 32; m <<= 1) a += __shfl_xor(a, m, 32);
      if (c == 0) cp[h] = a;
      else if (c == 1) cp[4 + h] = a;
      else if (c == 2) cd[h] = a;
      else cd[4 + h] = a;
    }
  }

  float av = 0.f, bv = fc1_b[lane];
#pragma unroll
  for (int k = 0; k < 8; ++k) {
    av += cp[k] * fc1_w[k * 32 + lane];
    bv += cd[k] * fc1_w[(8 + k) * 32 + lane];
  }
  Af[(size_t)bn * 32 + lane] = (_Float16)av;
  Bf[(size_t)bn * 32 + lane] = (_Float16)bv;
}

// ---------------------------------------------------------------------------
// K3: main MLP, transposed fc2 so fc3 needs only one cross-lane op.
//   yT(32x16) = W2^T (A operand, loop-invariant) @ x^T (B operand) + fc2_b (C)
// B-layout puts positions on lanes (col n = lane&15) and here uses contiguous
// feature range k in [0,16) for lanes<16, [16,32) for lanes>=16 -> one v16h
// load per row.  D-layout: lane = position, reg r = out-feature
// (r + 8*hi for c1, 16 + r + 8*hi for c2) -> lanes l and l+16 hold
// complementary feature halves of the SAME position:
//   out[j] = partial + shfl_xor(partial, 16).
// Two i-rows per iteration: 4 WMMAs, lo half stores row i, hi half row i+1.
// ---------------------------------------------------------------------------
__global__ __launch_bounds__(32) void k3_mlp(
    const _Float16* __restrict__ Af, const _Float16* __restrict__ Bf,
    const float* __restrict__ fc2_w, const float* __restrict__ fc2_b,
    const float* __restrict__ fc3_w, const float* __restrict__ fc3_b,
    float* __restrict__ out) {
  int ic = blockIdx.x & 15;
  int jt = (blockIdx.x >> 4) & 31;
  int b = blockIdx.x >> 9;
  int lane = threadIdx.x;
  int nlo = lane & 15;
  int hi = lane >> 4;  // 0 or 1

  // A operands: W2^T.  A1[m][k] = fc2_w[k][m] (m = nlo), A2[m][k] = fc2_w[k][16+m].
  // A k order used here, same as B: halves t=0..15 -> K = k0+t,
  // with k0 = 0 (lanes<16) / 16 (lanes>=16).
  v16h a1, a2;
  int k0 = hi ? 16 : 0;
#pragma unroll
  for (int t = 0; t < 16; ++t) {
    int Kt = k0 + t;
    a1[t] = (_Float16)fc2_w[Kt * 32 + nlo];
    a2[t] = (_Float16)fc2_w[Kt * 32 + 16 + nlo];
  }

  // per-reg constants: c1 reg r -> feature m = hi*8 + r; c2 -> 16 + hi*8 + r
  // fc2 bias goes straight into the WMMA C operand.
  v8f cb1, cb2;
  float w3a[8], w3b[8];
#pragma unroll
  for (int r = 0; r < 8; ++r) {
    cb1[r] = fc2_b[hi * 8 + r];
    cb2[r] = fc2_b[16 + hi * 8 + r];
    w3a[r] = fc3_w[hi * 8 + r];
    w3b[r] = fc3_w[16 + hi * 8 + r];
  }
  float b3 = fc3_b[0];

  // Bf tile: B-operand lane needs features kb..kb+15 (contiguous f16)
  int kb = hi ? 16 : 0;
  int j = jt * 16 + nlo;
  int jc = j < Ntot ? j : (Ntot - 1);
  v16h bfv = *(const v16h*)(Bf + ((size_t)b * Ntot + jc) * 32 + kb);

  v16h zeroh = {};
  int i0 = ic * 32;
  int iend = (i0 + 32 < Ntot) ? (i0 + 32) : Ntot;
  for (int i = i0; i < iend; i += 2) {
    int i1 = (i + 1 < iend) ? (i + 1) : i;
    v16h af0 = *(const v16h*)(Af + ((size_t)b * Ntot + i) * 32 + kb);
    v16h af1 = *(const v16h*)(Af + ((size_t)b * Ntot + i1) * 32 + kb);

    // x = relu(A_i + B_j) in packed f16
    v16h x0 = __builtin_elementwise_max(af0 + bfv, zeroh);
    v16h x1 = __builtin_elementwise_max(af1 + bfv, zeroh);

    v8f c1 = __builtin_amdgcn_wmma_f32_16x16x32_f16(
        false, a1, false, x0, (short)0, cb1, false, false);
    v8f c2 = __builtin_amdgcn_wmma_f32_16x16x32_f16(
        false, a2, false, x0, (short)0, cb2, false, false);
    v8f c3 = __builtin_amdgcn_wmma_f32_16x16x32_f16(
        false, a1, false, x1, (short)0, cb1, false, false);
    v8f c4 = __builtin_amdgcn_wmma_f32_16x16x32_f16(
        false, a2, false, x1, (short)0, cb2, false, false);

    float p0 = 0.f, p1 = 0.f;
#pragma unroll
    for (int r = 0; r < 8; ++r) {
      p0 += fmaxf(c1[r], 0.f) * w3a[r];
      p0 += fmaxf(c2[r], 0.f) * w3b[r];
      p1 += fmaxf(c3[r], 0.f) * w3a[r];
      p1 += fmaxf(c4[r], 0.f) * w3b[r];
    }
    float t0 = p0 + __shfl_xor(p0, 16, 32);
    float t1 = p1 + __shfl_xor(p1, 16, 32);

    // lo half stores row i, hi half stores row i+1 (coalesced 32-lane store)
    int row = hi ? i1 : i;
    float val = (hi ? t1 : t0) + b3;
    bool ok = (j < Ntot) && (hi ? (i + 1 < iend) : true);
    if (ok) out[((size_t)b * Ntot + row) * Ntot + j] = val;
  }
}

extern "C" void kernel_launch(void* const* d_in, const int* in_sizes, int n_in,
                              void* d_out, int out_size, void* d_ws, size_t ws_size,
                              hipStream_t stream) {
  const float* h_hat   = (const float*)d_in[0];
  const int*   pos_p   = (const int*)d_in[1];
  const int*   pos_d   = (const int*)d_in[2];
  const int*   solution= (const int*)d_in[3];
  const float* Wq1     = (const float*)d_in[4];
  const float* Wk1     = (const float*)d_in[5];
  const float* Wq2     = (const float*)d_in[6];
  const float* Wk2     = (const float*)d_in[7];
  const float* fc1_w   = (const float*)d_in[8];
  const float* fc1_b   = (const float*)d_in[9];
  const float* fc2_w   = (const float*)d_in[10];
  const float* fc2_b   = (const float*)d_in[11];
  const float* fc3_w   = (const float*)d_in[12];
  const float* fc3_b   = (const float*)d_in[13];
  float* out = (float*)d_out;

  float*     Vv = (float*)d_ws;                           // 4*B*H*D f32
  _Float16*  Af = (_Float16*)(Vv + 4 * Bsz * Hh * Dd);    // B*N*32 f16
  _Float16*  Bf = Af + (size_t)Bsz * Ntot * 32;           // B*N*32 f16

  k1_compat_vec<<<4 * Bsz * Hh, 128, 0, stream>>>(h_hat, pos_p, pos_d,
                                                  Wq1, Wk1, Wq2, Wk2, Vv);
  k2_feat<<<Bsz * Ntot, 32, 0, stream>>>(h_hat, solution, Vv, fc1_w, fc1_b,
                                         Af, Bf);
  k3_mlp<<<Bsz * 32 * 16, 32, 0, stream>>>(Af, Bf, fc2_w, fc2_b, fc3_w, fc3_b,
                                           out);
}
